// MetaTinyHAR_75299366633776
// MI455X (gfx1250) — compile-verified, matrix operand order checked by reading, not run on hardware
//
#include <hip/hip_runtime.h>
#include <hip/hip_bf16.h>
#include <cstdint>
#include <cstddef>

// ---------------------------------------------------------------------------
// CDNA5 (gfx1250) wave32 WMMA types & fragment loaders
// ---------------------------------------------------------------------------
typedef __attribute__((ext_vector_type(16))) _Float16 v16h;
typedef __attribute__((ext_vector_type(8)))  float    v8f;
typedef __attribute__((ext_vector_type(8)))  _Float16 h8;

union H16 { v16h v; h8 h[2]; };

// Load a 16x32 f16 A/B fragment from a row-major matrix (ld in elements).
// ISA layout (CDNA5 ISA): lanes 0-15 row m=lane hold K {0..7,16..23},
// lanes 16-31 row m=lane-16 hold K {8..15,24..31}.  Each lane: two 16B loads.
__device__ __forceinline__ v16h load_frag(const _Float16* p0, int ld, int kb) {
  int lane = threadIdx.x & 31;
  const _Float16* p = p0 + (size_t)(lane & 15) * ld + kb + ((lane >> 4) << 3);
  H16 u;
  u.h[0] = *(const h8*)(p);
  u.h[1] = *(const h8*)(p + 16);
  return u.v;
}

__device__ __forceinline__ float sigm(float x) { return 1.0f / (1.0f + expf(-x)); }

// ---------------------------------------------------------------------------
// Generic WMMA GEMM:  out(M,N) = act(A(M,K) @ W(N,K)^T + bias)
// A, W row-major f16, K multiple of 32, M multiple of 16, N multiple of 32.
// One wave per 16x32 output tile (two WMMAs share one A fragment).
// ---------------------------------------------------------------------------
template<bool RELU, bool OUT16>
__global__ __launch_bounds__(256) void gemm_wmma(
    const _Float16* __restrict__ A, int lda,
    const _Float16* __restrict__ W, int ldw,
    const float* __restrict__ bias,
    void* __restrict__ outv, int ldo,
    int K, int mTiles, int nPairs) {
  int wave = threadIdx.x >> 5;
  int lane = threadIdx.x & 31;
  int tile = blockIdx.x * 8 + wave;
  if (tile >= mTiles * nPairs) return;          // whole-wave uniform exit
  int mT = tile / nPairs, nP = tile - mT * nPairs;
  const _Float16* Ab  = A + (size_t)mT * 16 * lda;
  const _Float16* Wb0 = W + (size_t)nP * 32 * ldw;
  const _Float16* Wb1 = Wb0 + (size_t)16 * ldw;
  __builtin_prefetch(Wb0, 0, 0);                // global_prefetch_b8 (L2-resident)
  __builtin_prefetch(Wb1, 0, 0);
  v8f acc0 = {}, acc1 = {};
  for (int kb = 0; kb < K; kb += 32) {
    v16h a  = load_frag(Ab, lda, kb);
    v16h b0 = load_frag(Wb0, ldw, kb);
    v16h b1 = load_frag(Wb1, ldw, kb);
    acc0 = __builtin_amdgcn_wmma_f32_16x16x32_f16(false, a, false, b0,
                                                  (short)0, acc0, false, false);
    acc1 = __builtin_amdgcn_wmma_f32_16x16x32_f16(false, a, false, b1,
                                                  (short)0, acc1, false, false);
  }
  int n0 = nP * 32 + (lane & 15);
  int hl = lane >> 4;
  float bv0 = bias ? bias[n0] : 0.f;
  float bv1 = bias ? bias[n0 + 16] : 0.f;
  for (int r = 0; r < 8; ++r) {
    int m = mT * 16 + r + (hl << 3);
    float v0 = acc0[r] + bv0;
    float v1 = acc1[r] + bv1;
    if (RELU) { v0 = v0 > 0.f ? v0 : 0.f; v1 = v1 > 0.f ? v1 : 0.f; }
    if (OUT16) {
      ((_Float16*)outv)[(size_t)m * ldo + n0]      = (_Float16)v0;
      ((_Float16*)outv)[(size_t)m * ldo + n0 + 16] = (_Float16)v1;
    } else {
      ((float*)outv)[(size_t)m * ldo + n0]      = v0;
      ((float*)outv)[(size_t)m * ldo + n0 + 16] = v1;
    }
  }
}

// ---------------------------------------------------------------------------
// Per-sample implicit-GEMM conv1d (stride 2, pad 2, K=5) with fused
// base+LoRA weight (weff, rows padded to 32), bias + ReLU epilogue.
// Block: 256 thr = 8 waves, 64 output rows (c,t), im2col patch in LDS.
// Fully templated so all div/mod lower to shifts & magic multiplies.
// ---------------------------------------------------------------------------
template<int INC, int TIN, int KP>
__global__ __launch_bounds__(256) void conv_wmma_kernel(
    const _Float16* __restrict__ cur,   // (B*12, INC, TIN) f16
    const _Float16* __restrict__ weff,  // (B, 32, KP) f16
    const float* __restrict__ cbias,    // (28)
    _Float16* __restrict__ outp) {      // (B*12, 28, TOUT) f16
  constexpr int TOUT = TIN / 2;
  constexpr int KC = INC * 5;
  __shared__ __align__(16) _Float16 patch[64 * KP];
  int b = blockIdx.y;
  int tid = threadIdx.x;
  int rowBase = blockIdx.x * 64;
  for (int idx = tid; idx < 64 * KP; idx += 256) {
    int row = idx / KP, kk = idx - row * KP;
    _Float16 val = (_Float16)0.f;
    if (kk < KC) {
      int ic = kk / 5, k = kk - ic * 5;
      int grow = rowBase + row;
      int c = grow / TOUT, t = grow - c * TOUT;   // TOUT pow2 -> shift
      int ti = 2 * t + k - 2;
      if (ti >= 0 && ti < TIN)
        val = cur[((size_t)(b * 12 + c) * INC + ic) * TIN + ti];
    }
    patch[row * KP + kk] = val;
  }
  __syncthreads();
  int wave = tid >> 5, lane = tid & 31;
  int mSub = wave >> 1, nT = wave & 1;
  const _Float16* Ab = patch + mSub * 16 * KP;
  const _Float16* Wb = weff + ((size_t)b * 32 + nT * 16) * KP;
  __builtin_prefetch(Wb, 0, 0);
  v8f acc = {};
  for (int kb = 0; kb < KP; kb += 32) {
    v16h a  = load_frag(Ab, KP, kb);
    v16h bf = load_frag(Wb, KP, kb);
    acc = __builtin_amdgcn_wmma_f32_16x16x32_f16(false, a, false, bf,
                                                 (short)0, acc, false, false);
  }
  int o = nT * 16 + (lane & 15);
  if (o < 28) {
    float bv = cbias[o];
    int hl = lane >> 4;
    for (int r = 0; r < 8; ++r) {
      int grow = rowBase + mSub * 16 + r + (hl << 3);
      int c = grow / TOUT, t = grow - c * TOUT;
      float v = acc[r] + bv;
      v = v > 0.f ? v : 0.f;
      outp[((size_t)(b * 12 + c) * 28 + o) * TOUT + t] = (_Float16)v;
    }
  }
}

// GroupNorm in place: one block per (row, group); 7 features x TT elements.
template<int TT>
__global__ __launch_bounds__(128) void groupnorm_ip(
    _Float16* __restrict__ buf, const float* __restrict__ g,
    const float* __restrict__ bta) {
  __shared__ float s1[128], s2[128];
  int row = blockIdx.x, grp = blockIdx.y, tid = threadIdx.x;
  _Float16* base = buf + ((size_t)row * 28 + grp * 7) * TT;
  constexpr int N = 7 * TT;
  float a = 0.f, q = 0.f;
  for (int i = tid; i < N; i += 128) { float x = (float)base[i]; a += x; q += x * x; }
  s1[tid] = a; s2[tid] = q; __syncthreads();
  for (int s = 64; s > 0; s >>= 1) {
    if (tid < s) { s1[tid] += s1[tid + s]; s2[tid] += s2[tid + s]; }
    __syncthreads();
  }
  float mean = s1[0] * (1.f / N);
  float var  = s2[0] * (1.f / N) - mean * mean;
  float rstd = rsqrtf(var + 1e-5f);
  for (int i = tid; i < N; i += 128) {
    int f = grp * 7 + i / TT;                     // TT pow2 -> shift
    float x = ((float)base[i] - mean) * rstd * g[f] + bta[f];
    base[i] = (_Float16)x;
  }
}

// ---------------------------------------------------------------------------
// Small VALU kernels: packing, attention, layernorm, LSTM, pooling
// ---------------------------------------------------------------------------
__global__ void pack_w16(const float* __restrict__ src, _Float16* __restrict__ dst,
                         int rows, int K, int rowsPad, int Kpad) {
  int idx = blockIdx.x * blockDim.x + threadIdx.x;
  int total = rowsPad * Kpad;
  if (idx >= total) return;
  int r = idx / Kpad, k = idx - r * Kpad;
  float v = (r < rows && k < K) ? src[(size_t)r * K + k] : 0.f;
  dst[idx] = (_Float16)v;
}

__global__ void pack_bias(const float* __restrict__ src, float* __restrict__ dst,
                          int n, int npad) {
  int i = blockIdx.x * blockDim.x + threadIdx.x;
  if (i < npad) dst[i] = (i < n) ? src[i] : 0.f;
}

__global__ void add_bias2(const float* a, const float* b, float* dst, int n) {
  int i = blockIdx.x * blockDim.x + threadIdx.x;
  if (i < n) dst[i] = a[i] + b[i];
}

// x (256,512,12) f32 -> curr0 (3072,1,512) f16
__global__ void pack_x(const float* __restrict__ x, _Float16* __restrict__ dst) {
  int idx = blockIdx.x * blockDim.x + threadIdx.x;
  if (idx >= 3072 * 512) return;
  int row = idx >> 9, t = idx & 511;
  int b = row / 12, c = row - b * 12;
  dst[idx] = (_Float16)x[((size_t)b * 512 + t) * 12 + c];
}

// [z | layer_emb[i]] -> (256, 96) f16
__global__ void pack_zle(const float* __restrict__ z, const float* __restrict__ le,
                         _Float16* __restrict__ dst) {
  int idx = blockIdx.x * blockDim.x + threadIdx.x;
  if (idx >= 256 * 96) return;
  int b = idx / 96, c = idx - b * 96;
  dst[idx] = (_Float16)((c < 64) ? z[b * 64 + c] : le[c - 64]);
}

// W_eff[b,o,kk] = conv_w[o,kk] + scale * sum_r Bm[b,o,r] * A[b,r,kk]
template<int ICK, int LDA, int KP>
__global__ void build_weff(const float* __restrict__ Aout, const float* __restrict__ Bout,
                           const float* __restrict__ cw, _Float16* __restrict__ weff,
                           float scale) {
  int idx = blockIdx.x * blockDim.x + threadIdx.x;
  if (idx >= 256 * 32 * KP) return;
  int b = idx / (32 * KP);
  int rem = idx - b * 32 * KP;
  int o = rem / KP, kk = rem - o * KP;
  float v = 0.f;
  if (o < 28 && kk < ICK) {
    float s = 0.f;
    for (int r = 0; r < 8; ++r)
      s += Bout[(size_t)b * 224 + o * 8 + r] * Aout[(size_t)b * LDA + r * ICK + kk];
    v = cw[o * ICK + kk] + scale * s;
  }
  weff[idx] = (_Float16)v;
}

// curr4 (3072,28,32) -> t_in (98304, 32-pad) f16
__global__ void pack_tin(const _Float16* __restrict__ cur4, _Float16* __restrict__ tin) {
  int idx = blockIdx.x * blockDim.x + threadIdx.x;
  if (idx >= 98304 * 32) return;
  int row = idx >> 5, f = idx & 31;
  int b = row / 384, rem = row - b * 384;
  int t = rem / 12, c = rem - t * 12;
  _Float16 v = (_Float16)0.f;
  if (f < 28) v = cur4[(((size_t)(b * 12 + c)) * 28 + f) * 32 + t];
  tin[idx] = v;
}

// Multi-head attention over 12 sensors, head_dim 7.  qkv (98304,96) f16.
__global__ void attn_heads(const _Float16* __restrict__ qkv, _Float16* __restrict__ ao) {
  int tid = blockIdx.x * blockDim.x + threadIdx.x;
  if (tid >= 8192 * 48) return;
  int n = tid / 48, rem = tid - n * 48;
  int h = rem / 12, s = rem - h * 12;
  const _Float16* Q = qkv + ((size_t)(n * 12 + s)) * 96 + h * 7;
  float q[7];
  for (int d = 0; d < 7; ++d) q[d] = (float)Q[d];
  float sc[12], mx = -1e30f;
  const float scale = 0.3779644730092272f;  // 1/sqrt(7)
  for (int t = 0; t < 12; ++t) {
    const _Float16* Kp = qkv + ((size_t)(n * 12 + t)) * 96 + 28 + h * 7;
    float d0 = 0.f;
    for (int d = 0; d < 7; ++d) d0 += q[d] * (float)Kp[d];
    sc[t] = d0 * scale;
    mx = fmaxf(mx, sc[t]);
  }
  float sum = 0.f;
  for (int t = 0; t < 12; ++t) { sc[t] = expf(sc[t] - mx); sum += sc[t]; }
  float inv = 1.f / sum;
  _Float16* O = ao + ((size_t)(n * 12 + s)) * 32 + h * 7;
  for (int d = 0; d < 7; ++d) {
    float a = 0.f;
    for (int t = 0; t < 12; ++t)
      a += sc[t] * (float)qkv[((size_t)(n * 12 + t)) * 96 + 56 + h * 7 + d];
    O[d] = (_Float16)(a * inv);
  }
  if (h == 3) {
    _Float16* P = ao + ((size_t)(n * 12 + s)) * 32;
    P[28] = P[29] = P[30] = P[31] = (_Float16)0.f;
  }
}

// x1 = LN(t_in + res) over 28 features -> f16 (ld 32, zero pad)
__global__ void ln_res(const _Float16* __restrict__ tin, const float* __restrict__ res,
                       const float* __restrict__ g, const float* __restrict__ bb,
                       _Float16* __restrict__ outp) {
  int row = blockIdx.x * blockDim.x + threadIdx.x;
  if (row >= 98304) return;
  float v[28]; float m = 0.f;
  for (int f = 0; f < 28; ++f) {
    v[f] = (float)tin[(size_t)row * 32 + f] + res[(size_t)row * 32 + f];
    m += v[f];
  }
  m *= (1.f / 28.f);
  float var = 0.f;
  for (int f = 0; f < 28; ++f) { float d = v[f] - m; var += d * d; }
  var *= (1.f / 28.f);
  float rstd = rsqrtf(var + 1e-5f);
  _Float16* o = outp + (size_t)row * 32;
  for (int f = 0; f < 28; ++f) o[f] = (_Float16)((v[f] - m) * rstd * g[f] + bb[f]);
  o[28] = o[29] = o[30] = o[31] = (_Float16)0.f;
}

// x2 = LN(x1 + res); write packed into fusion input fin (8192, 352-pad)
__global__ void ln_res_fin(const _Float16* __restrict__ x1, const float* __restrict__ res,
                           const float* __restrict__ g, const float* __restrict__ bb,
                           _Float16* __restrict__ fin) {
  int row = blockIdx.x * blockDim.x + threadIdx.x;
  if (row >= 98304) return;
  float v[28]; float m = 0.f;
  for (int f = 0; f < 28; ++f) {
    v[f] = (float)x1[(size_t)row * 32 + f] + res[(size_t)row * 32 + f];
    m += v[f];
  }
  m *= (1.f / 28.f);
  float var = 0.f;
  for (int f = 0; f < 28; ++f) { float d = v[f] - m; var += d * d; }
  var *= (1.f / 28.f);
  float rstd = rsqrtf(var + 1e-5f);
  int n = row / 12, c = row - n * 12;
  _Float16* o = fin + (size_t)n * 352 + c * 28;
  for (int f = 0; f < 28; ++f) o[f] = (_Float16)((v[f] - m) * rstd * g[f] + bb[f]);
  if (c == 11) {
    _Float16* p = fin + (size_t)n * 352 + 336;
    for (int j = 0; j < 16; ++j) p[j] = (_Float16)0.f;
  }
}

// LSTM: 16 samples / block, WMMA across the batch dim for h @ whh^T.
// xg already holds x@wih^T + bih + bhh.  whh f16 (224, 64-pad).
__global__ __launch_bounds__(64) void lstm_wmma(const float* __restrict__ xg,
                                                const _Float16* __restrict__ whh,
                                                float* __restrict__ lo) {
  __shared__ __align__(16) _Float16 hsh[16 * 64];
  __shared__ float csh[16 * 56];
  __shared__ float gsh[16 * 224];
  int tid = threadIdx.x;
  int s0 = blockIdx.x * 16;
  for (int i = tid; i < 16 * 64; i += 64) hsh[i] = (_Float16)0.f;
  for (int i = tid; i < 16 * 56; i += 64) csh[i] = 0.f;
  __syncthreads();
  int wave = tid >> 5, lane = tid & 31;
  int nlo = lane & 15, hl = lane >> 4;
  for (int t = 0; t < 32; ++t) {
    v16h ha0 = load_frag(hsh, 64, 0);
    v16h ha1 = load_frag(hsh, 64, 32);
    for (int qn = 0; qn < 7; ++qn) {
      int nt = wave * 7 + qn;
      int n = nt * 16 + nlo;
      v8f acc;
      for (int r = 0; r < 8; ++r) {
        int m = r + (hl << 3);
        acc[r] = xg[((size_t)(s0 + m) * 32 + t) * 224 + n];
      }
      const _Float16* Wb = whh + (size_t)nt * 16 * 64;
      v16h b0 = load_frag(Wb, 64, 0);
      v16h b1 = load_frag(Wb, 64, 32);
      acc = __builtin_amdgcn_wmma_f32_16x16x32_f16(false, ha0, false, b0,
                                                   (short)0, acc, false, false);
      acc = __builtin_amdgcn_wmma_f32_16x16x32_f16(false, ha1, false, b1,
                                                   (short)0, acc, false, false);
      for (int r = 0; r < 8; ++r) gsh[(r + (hl << 3)) * 224 + n] = acc[r];
    }
    __syncthreads();
    for (int i = tid; i < 16 * 56; i += 64) {
      int m = i / 56, u = i - m * 56;
      float gi = gsh[m * 224 + u];
      float gf = gsh[m * 224 + 56 + u];
      float gg = gsh[m * 224 + 112 + u];
      float go = gsh[m * 224 + 168 + u];
      float c = sigm(gf) * csh[i] + sigm(gi) * tanhf(gg);
      float h = sigm(go) * tanhf(c);
      csh[i] = c;
      hsh[m * 64 + u] = (_Float16)h;
      lo[((size_t)(s0 + m) * 32 + t) * 56 + u] = h;
    }
    __syncthreads();
  }
}

// attention pooling + classifier, one block per sample
__global__ __launch_bounds__(64) void pool_cls(const float* __restrict__ lo,
    const float* __restrict__ attw, const float* __restrict__ attb,
    const float* __restrict__ gamma, const float* __restrict__ clsw,
    const float* __restrict__ clsb, float* __restrict__ outp) {
  __shared__ float e[32];
  __shared__ float fvec[56];
  int b = blockIdx.x, tid = threadIdx.x;
  if (tid < 32) {
    float d = 0.f;
    for (int u = 0; u < 56; ++u) d += lo[((size_t)b * 32 + tid) * 56 + u] * attw[u];
    e[tid] = tanhf(d + attb[0]);
  }
  __syncthreads();
  if (tid == 0) {
    float mx = -1e30f;
    for (int t = 0; t < 32; ++t) mx = fmaxf(mx, e[t]);
    float s = 0.f;
    for (int t = 0; t < 32; ++t) { e[t] = expf(e[t] - mx); s += e[t]; }
    float inv = 1.f / s;
    for (int t = 0; t < 32; ++t) e[t] *= inv;
  }
  __syncthreads();
  if (tid < 56) {
    float c = 0.f;
    for (int t = 0; t < 32; ++t) c += lo[((size_t)b * 32 + t) * 56 + tid] * e[t];
    fvec[tid] = lo[((size_t)b * 32 + 31) * 56 + tid] + gamma[0] * c;
  }
  __syncthreads();
  if (tid < 10) {
    float d = clsb[tid];
    for (int u = 0; u < 56; ++u) d += fvec[u] * clsw[tid * 56 + u];
    outp[b * 10 + tid] = d;
  }
}

// ---------------------------------------------------------------------------
// Host orchestration
// ---------------------------------------------------------------------------
extern "C" void kernel_launch(void* const* d_in, const int* in_sizes, int n_in,
                              void* d_out, int out_size, void* d_ws, size_t ws_size,
                              hipStream_t stream) {
  (void)in_sizes; (void)n_in; (void)out_size; (void)ws_size;

  const float* x         = (const float*)d_in[0];
  const float* z         = (const float*)d_in[1];
  const float* layer_emb = (const float*)d_in[2];
  const float* mlp_w1    = (const float*)d_in[3];
  const float* mlp_b1    = (const float*)d_in[4];
  const float* mlp_w2    = (const float*)d_in[5];
  const float* mlp_b2    = (const float*)d_in[6];
  const int iAW[4] = {7, 11, 15, 19}, iABv[4] = {8, 12, 16, 20};
  const int iBW[4] = {9, 13, 17, 21}, iBBv[4] = {10, 14, 18, 22};
  const int iCW[4] = {23, 27, 31, 35}, iCB[4] = {24, 28, 32, 36};
  const int iGG[4] = {25, 29, 33, 37}, iGB[4] = {26, 30, 34, 38};
  const float* attn_wi = (const float*)d_in[39];
  const float* attn_bi = (const float*)d_in[40];
  const float* attn_wo = (const float*)d_in[41];
  const float* attn_bo = (const float*)d_in[42];
  const float* ln1_g   = (const float*)d_in[43];
  const float* ln1_b   = (const float*)d_in[44];
  const float* ff_w1   = (const float*)d_in[45];
  const float* ff_b1   = (const float*)d_in[46];
  const float* ff_w2   = (const float*)d_in[47];
  const float* ff_b2   = (const float*)d_in[48];
  const float* ln2_g   = (const float*)d_in[49];
  const float* ln2_b   = (const float*)d_in[50];
  const float* fus_w   = (const float*)d_in[51];
  const float* fus_b   = (const float*)d_in[52];
  const float* wih     = (const float*)d_in[53];
  const float* bih     = (const float*)d_in[54];
  const float* whh     = (const float*)d_in[55];
  const float* bhh     = (const float*)d_in[56];
  const float* att_w   = (const float*)d_in[57];
  const float* att_b   = (const float*)d_in[58];
  const float* gamma   = (const float*)d_in[59];
  const float* cls_w   = (const float*)d_in[60];
  const float* cls_b   = (const float*)d_in[61];
  float* out = (float*)d_out;

  char* wsb = (char*)d_ws;
  size_t off = 0;
  auto alloc = [&](size_t bytes) -> void* {
    void* p = wsb + off;
    off += (bytes + 255) & ~(size_t)255;
    return p;
  };
  auto cdiv = [](int a, int b) { return (a + b - 1) / b; };

  // f16 padded weights
  _Float16* w_mlp1 = (_Float16*)alloc(128 * 96 * 2);
  _Float16* w_mlp2 = (_Float16*)alloc(128 * 128 * 2);
  _Float16* w_aw   = (_Float16*)alloc(1120 * 128 * 2);  // reused per layer
  _Float16* w_bw   = (_Float16*)alloc(224 * 128 * 2);   // reused per layer
  _Float16* w_wi   = (_Float16*)alloc(96 * 32 * 2);
  _Float16* w_wo   = (_Float16*)alloc(32 * 32 * 2);
  _Float16* w_f1   = (_Float16*)alloc(64 * 32 * 2);
  _Float16* w_f2   = (_Float16*)alloc(32 * 64 * 2);
  _Float16* w_fus  = (_Float16*)alloc(64 * 352 * 2);
  _Float16* w_wih  = (_Float16*)alloc(224 * 64 * 2);
  _Float16* w_whh  = (_Float16*)alloc(224 * 64 * 2);
  // padded f32 biases
  float* abPad = (float*)alloc(64 * 4);
  float* biP   = (float*)alloc(96 * 4);
  float* boP   = (float*)alloc(32 * 4);
  float* f1P   = (float*)alloc(64 * 4);
  float* f2P   = (float*)alloc(32 * 4);
  float* fusP  = (float*)alloc(64 * 4);
  float* bb2   = (float*)alloc(224 * 4);
  // activations
  _Float16* zle  = (_Float16*)alloc(256 * 96 * 2);
  _Float16* h1   = (_Float16*)alloc(256 * 128 * 2);
  _Float16* h2   = (_Float16*)alloc(256 * 128 * 2);
  float*    Aout = (float*)alloc((size_t)256 * 1120 * 4);
  float*    Bout = (float*)alloc((size_t)256 * 224 * 4);
  _Float16* weff = (_Float16*)alloc((size_t)256 * 32 * 160 * 2);
  _Float16* bufA = (_Float16*)alloc((size_t)3072 * 28 * 128 * 2);  // 22 MB
  _Float16* bufB = (_Float16*)alloc((size_t)3072 * 28 * 256 * 2);  // 44 MB
  _Float16* tin  = (_Float16*)alloc((size_t)98304 * 32 * 2);
  _Float16* qkv  = (_Float16*)alloc((size_t)98304 * 96 * 2);
  _Float16* ao   = (_Float16*)alloc((size_t)98304 * 32 * 2);
  float*    resid= (float*)alloc((size_t)98304 * 32 * 4);
  _Float16* x1b  = (_Float16*)alloc((size_t)98304 * 32 * 2);
  _Float16* ffh  = (_Float16*)alloc((size_t)98304 * 64 * 2);
  _Float16* finb = (_Float16*)alloc((size_t)8192 * 352 * 2);
  _Float16* seq  = (_Float16*)alloc((size_t)8192 * 64 * 2);
  float*    xg   = (float*)alloc((size_t)8192 * 224 * 4);
  float*    lo   = (float*)alloc((size_t)8192 * 56 * 4);

  // ---- static weight packing ----
  pack_w16<<<cdiv(128 * 96, 256), 256, 0, stream>>>(mlp_w1, w_mlp1, 128, 96, 128, 96);
  pack_w16<<<cdiv(128 * 128, 256), 256, 0, stream>>>(mlp_w2, w_mlp2, 128, 128, 128, 128);
  pack_w16<<<cdiv(96 * 32, 256), 256, 0, stream>>>(attn_wi, w_wi, 84, 28, 96, 32);
  pack_w16<<<cdiv(32 * 32, 256), 256, 0, stream>>>(attn_wo, w_wo, 28, 28, 32, 32);
  pack_w16<<<cdiv(64 * 32, 256), 256, 0, stream>>>(ff_w1, w_f1, 56, 28, 64, 32);
  pack_w16<<<cdiv(32 * 64, 256), 256, 0, stream>>>(ff_w2, w_f2, 28, 56, 32, 64);
  pack_w16<<<cdiv(64 * 352, 256), 256, 0, stream>>>(fus_w, w_fus, 56, 336, 64, 352);
  pack_w16<<<cdiv(224 * 64, 256), 256, 0, stream>>>(wih, w_wih, 224, 56, 224, 64);
  pack_w16<<<cdiv(224 * 64, 256), 256, 0, stream>>>(whh, w_whh, 224, 56, 224, 64);
  pack_bias<<<1, 96, 0, stream>>>(attn_bi, biP, 84, 96);
  pack_bias<<<1, 32, 0, stream>>>(attn_bo, boP, 28, 32);
  pack_bias<<<1, 64, 0, stream>>>(ff_b1, f1P, 56, 64);
  pack_bias<<<1, 32, 0, stream>>>(ff_b2, f2P, 28, 32);
  pack_bias<<<1, 64, 0, stream>>>(fus_b, fusP, 56, 64);
  add_bias2<<<1, 224, 0, stream>>>(bih, bhh, bb2, 224);

  // ---- input transpose: x -> (3072,1,512) f16 ----
  pack_x<<<cdiv(3072 * 512, 256), 256, 0, stream>>>(x, bufA);

  // ---- LoRA conv stack ----
  const int NaP[4] = {64, 1120, 1120, 1120};   // LoRA-A padded width (mult of 32)
  const int NaR[4] = {40, 1120, 1120, 1120};
  _Float16* curIn = bufA;
  _Float16* curOut = bufB;
  for (int i = 0; i < 4; ++i) {
    const float* awp = (const float*)d_in[iAW[i]];
    const float* abp = (const float*)d_in[iABv[i]];
    const float* bwp = (const float*)d_in[iBW[i]];
    const float* bbp = (const float*)d_in[iBBv[i]];
    const float* cw  = (const float*)d_in[iCW[i]];
    const float* cb  = (const float*)d_in[iCB[i]];
    const float* gg  = (const float*)d_in[iGG[i]];
    const float* gb  = (const float*)d_in[iGB[i]];

    pack_w16<<<cdiv(NaP[i] * 128, 256), 256, 0, stream>>>(awp, w_aw, NaR[i], 128, NaP[i], 128);
    pack_w16<<<cdiv(224 * 128, 256), 256, 0, stream>>>(bwp, w_bw, 224, 128, 224, 128);
    const float* biasA = abp;
    if (i == 0) { pack_bias<<<1, 64, 0, stream>>>(abp, abPad, 40, 64); biasA = abPad; }

    // hypernetwork (WMMA GEMMs, 2 N-tiles per wave)
    pack_zle<<<cdiv(256 * 96, 256), 256, 0, stream>>>(z, layer_emb + i * 32, zle);
    gemm_wmma<true, true><<<cdiv(16 * 4, 8), 256, 0, stream>>>(
        zle, 96, w_mlp1, 96, mlp_b1, h1, 128, 96, 16, 4);
    gemm_wmma<true, true><<<cdiv(16 * 4, 8), 256, 0, stream>>>(
        h1, 128, w_mlp2, 128, mlp_b2, h2, 128, 128, 16, 4);
    gemm_wmma<false, false><<<cdiv(16 * (NaP[i] / 32), 8), 256, 0, stream>>>(
        h2, 128, w_aw, 128, biasA, Aout, NaP[i], 128, 16, NaP[i] / 32);
    gemm_wmma<false, false><<<cdiv(16 * 7, 8), 256, 0, stream>>>(
        h2, 128, w_bw, 128, bbp, Bout, 224, 128, 16, 7);

    // fused base+LoRA weight, conv, GroupNorm (all compile-time shapes)
    if (i == 0) {
      build_weff<5, 64, 32><<<cdiv(256 * 32 * 32, 256), 256, 0, stream>>>(
          Aout, Bout, cw, weff, 1.0f /* ALPHA/R */);
      conv_wmma_kernel<1, 512, 32><<<dim3(48, 256), 256, 0, stream>>>(curIn, weff, cb, curOut);
      groupnorm_ip<256><<<dim3(3072, 4), 128, 0, stream>>>(curOut, gg, gb);
    } else {
      build_weff<140, 1120, 160><<<cdiv(256 * 32 * 160, 256), 256, 0, stream>>>(
          Aout, Bout, cw, weff, 1.0f);
      if (i == 1) {
        conv_wmma_kernel<28, 256, 160><<<dim3(24, 256), 256, 0, stream>>>(curIn, weff, cb, curOut);
        groupnorm_ip<128><<<dim3(3072, 4), 128, 0, stream>>>(curOut, gg, gb);
      } else if (i == 2) {
        conv_wmma_kernel<28, 128, 160><<<dim3(12, 256), 256, 0, stream>>>(curIn, weff, cb, curOut);
        groupnorm_ip<64><<<dim3(3072, 4), 128, 0, stream>>>(curOut, gg, gb);
      } else {
        conv_wmma_kernel<28, 64, 160><<<dim3(6, 256), 256, 0, stream>>>(curIn, weff, cb, curOut);
        groupnorm_ip<32><<<dim3(3072, 4), 128, 0, stream>>>(curOut, gg, gb);
      }
    }
    _Float16* tmp = curIn; curIn = curOut; curOut = tmp;
  }
  // curIn == curr4 (3072, 28, 32)

  // ---- transformer over sensors ----
  pack_tin<<<cdiv(98304 * 32, 256), 256, 0, stream>>>(curIn, tin);
  gemm_wmma<false, true><<<cdiv(6144 * 3, 8), 256, 0, stream>>>(
      tin, 32, w_wi, 32, biP, qkv, 96, 32, 6144, 3);
  attn_heads<<<cdiv(8192 * 48, 256), 256, 0, stream>>>(qkv, ao);
  gemm_wmma<false, false><<<cdiv(6144 * 1, 8), 256, 0, stream>>>(
      ao, 32, w_wo, 32, boP, resid, 32, 32, 6144, 1);
  ln_res<<<cdiv(98304, 256), 256, 0, stream>>>(tin, resid, ln1_g, ln1_b, x1b);
  gemm_wmma<true, true><<<cdiv(6144 * 2, 8), 256, 0, stream>>>(
      x1b, 32, w_f1, 32, f1P, ffh, 64, 32, 6144, 2);
  gemm_wmma<false, false><<<cdiv(6144 * 1, 8), 256, 0, stream>>>(
      ffh, 64, w_f2, 64, f2P, resid, 32, 64, 6144, 1);
  ln_res_fin<<<cdiv(98304, 256), 256, 0, stream>>>(x1b, resid, ln2_g, ln2_b, finb);

  // ---- fusion + LSTM input gates ----
  gemm_wmma<false, true><<<cdiv(512 * 2, 8), 256, 0, stream>>>(
      finb, 352, w_fus, 352, fusP, seq, 64, 352, 512, 2);
  gemm_wmma<false, false><<<cdiv(512 * 7, 8), 256, 0, stream>>>(
      seq, 64, w_wih, 64, bb2, xg, 224, 64, 512, 7);

  // ---- LSTM recurrence (WMMA over batch) ----
  lstm_wmma<<<16, 64, 0, stream>>>(xg, w_whh, lo);

  // ---- attention pooling + classifier ----
  pool_cls<<<256, 64, 0, stream>>>(lo, att_w, att_b, gamma, cls_w, cls_b, out);
}
